// WithMemoryBuffer_64201171141021
// MI455X (gfx1250) — matrix operand order repaired and checked by reading, not compiled, not run on hardware
//
#include <hip/hip_runtime.h>
#include <math.h>

typedef _Float16 h16;
typedef __attribute__((ext_vector_type(16))) _Float16 v16h;
typedef __attribute__((ext_vector_type(8)))  _Float16 v8h;
typedef __attribute__((ext_vector_type(8)))  float    v8f;

namespace {
constexpr int kB = 16384;
constexpr int kT = 128;
constexpr int kH = 256;
constexpr int kOut = 10;
constexpr int kFT = kH / 16;     // 16 feature tiles (A/W rows, D rows)
constexpr int kKT = kH / 32;     // 8 K tiles
constexpr int kWaves = 4;
constexpr int kBlock = kWaves * 32;
constexpr int kColsPerWave = 16;               // batch columns per wave
constexpr int kBRows = kWaves * kColsPerWave;  // 64 batch rows per block
constexpr int kLut = 16;
constexpr float kEps = 1e-5f;

constexpr int kWBytes   = kH * kH * 2;     // 131072: pi-permuted W in A-fragment layout (f16)
constexpr int kLutBytes = kLut * kH * 2;   //   8192: tanh(v*ew+eb) lookup (f16)
constexpr int kOwBytes  = kOut * kH * 4;   //  10240: out_w (f32)
constexpr int kUbBytes  = kH * 4;          //   1024: update_b (f32, feeds WMMA C init)
constexpr int kGbBytes  = kH * 2;          //    512 each: gamma / beta (f16)
constexpr int kSmemBytes = kWBytes + kLutBytes + kOwBytes + kUbBytes + 2 * kGbBytes; // 151552
}  // namespace

__device__ __forceinline__ float fast_tanh(float x) {
#if __has_builtin(__builtin_amdgcn_tanhf)
  return __builtin_amdgcn_tanhf(x);   // v_tanh_f32 on gfx1250
#else
  return tanhf(x);
#endif
}

// Prescribe (2x DS-read, 1x WMMA) x16 to the machine scheduler for one K-tile:
// forces the next A fragment's ds_load_b128 pair between WMMAs, so the register
// allocator must double-buffer the A registers (kills the WAR serialization that
// produced s_wait_dscnt 0x0 before every WMMA). Lowers to zero instructions.
__device__ __forceinline__ void sched_ds_wmma_interleave() {
#if __has_builtin(__builtin_amdgcn_sched_group_barrier)
#pragma unroll
  for (int i = 0; i < 16; ++i) {
    __builtin_amdgcn_sched_group_barrier(0x100, 2, 0);  // 2 DS reads
    __builtin_amdgcn_sched_group_barrier(0x008, 1, 0);  // 1 MFMA/WMMA
  }
#endif
}

extern "C" __global__ __launch_bounds__(kBlock)
__attribute__((amdgpu_waves_per_eu(4)))   // cap 256 VGPRs: no vgpr_msb churn
void rnn_scan_wmma(const float* __restrict__ x,
                   const float* __restrict__ embed_w,
                   const float* __restrict__ embed_b,
                   const float* __restrict__ update_w,
                   const float* __restrict__ update_b,
                   const float* __restrict__ gamma,
                   const float* __restrict__ beta,
                   const float* __restrict__ out_w,
                   const float* __restrict__ out_b,
                   float* __restrict__ out) {
  extern __shared__ char smem[];
  h16*   Wl  = reinterpret_cast<h16*>(smem);
  h16*   lut = reinterpret_cast<h16*>(smem + kWBytes);
  float* owl = reinterpret_cast<float*>(smem + kWBytes + kLutBytes);
  float* ubl = reinterpret_cast<float*>(smem + kWBytes + kLutBytes + kOwBytes);
  h16*   glh = reinterpret_cast<h16*>(smem + kWBytes + kLutBytes + kOwBytes + kUbBytes);
  h16*   blh = glh + kH;

  const int tid  = threadIdx.x;
  const int wave = tid >> 5;
  const int lane = tid & 31;
  const int m    = lane & 15;   // batch column owned by this lane
  const int dh   = lane >> 4;   // which feature half this lane holds in C/D

  // ---------- init LDS (read-only thereafter) ----------
  // embed lookup: inp[v][f] = tanh(v*ew[f] + eb[f]), v = 0..15, by TRUE feature index
  for (int i = tid; i < kLut * kH; i += kBlock) {
    int v = i / kH, f = i % kH;
    lut[i] = (h16)fast_tanh((float)v * embed_w[f] + embed_b[f]);
  }
  // W in A-fragment layout with K permuted by pi so the D->B repack is lane-local:
  //   pi(32kt + 16hi + lo) = 32kt + 16*(lo>>3) + 8*hi + (lo&7)
  // A-layout (16-bit 16x32): lane L row M=L%16; half j -> Klocal = (j<8?j:j+8) + 8*(L>=16)
  for (int i = tid; i < kH * kH; i += kBlock) {
    int f = i >> 9;            // fragment: 32 lanes * 16 halves
    int r = i & 511;
    int L = r >> 4;
    int j = r & 15;
    int kt = f >> 4, ft = f & 15;
    int row = ft * 16 + (L & 15);
    int Klocal = (j < 8 ? j : j + 8) + ((L >> 4) << 3);
    int kperm = kt * 32 + Klocal;
    Wl[i] = (h16)update_w[row * kH + kperm];
  }
  for (int i = tid; i < kOut * kH; i += kBlock) owl[i] = out_w[i];
  for (int i = tid; i < kH; i += kBlock) {
    ubl[i] = update_b[i];
    glh[i] = (h16)gamma[i];
    blh[i] = (h16)beta[i];
  }
  __syncthreads();

  const int rowBase = blockIdx.x * kBRows + wave * kColsPerWave;
  const float* xptr = x + (size_t)(rowBase + m) * kT;

  // register-resident hidden state, packed f16 in B-fragment order (64 VGPRs):
  // ps[kt][j] = h[permuted-K feature 32kt + 16*(j>>3) + 8*dh + (j&7)][column m]
  v16h ps[kKT];
#pragma unroll
  for (int kt = 0; kt < kKT; ++kt) {
    v16h z;
#pragma unroll
    for (int j = 0; j < 16; ++j) z[j] = (h16)0.0f;
    ps[kt] = z;
  }

  float xv = xptr[0];   // x value for step t, preloaded one step ahead

  for (int t = 0; t < kT; ++t) {
    // Compiler-level memory clobber: stops LICM from hoisting the (loop-invariant)
    // LDS W-fragment loads out of the scan, which would force scratch spills under
    // the 256-VGPR cap. Emits no instruction; ps/acc stay in registers (SSA).
    asm volatile("" ::: "memory");

    __builtin_prefetch(xptr + t + 16, 0, 0);    // global_prefetch_b8 on the x stream
    const float xv_next = xptr[(t + 1) & (kT - 1)];  // consumed NEXT step: load latency
                                                     // hidden behind the whole timestep
    int xi = (int)xv;
    xi = xi < 0 ? 0 : (xi > kLut - 1 ? kLut - 1 : xi);
    const h16* lrow = lut + xi * kH;

    // C initialized with update_b (feature-indexed) -> bias folded into the WMMA chain
    v8f acc[kFT];
#pragma unroll
    for (int ft = 0; ft < kFT; ++ft)
      acc[ft] = *(const v8f*)(ubl + ft * 16 + 8 * dh);

#pragma unroll
    for (int kt = 0; kt < kKT; ++kt) {
      // B fragment: activations = h_prev + embed(x_t); ps is already in B order
      const h16* lp = lrow + kt * 32 + 8 * dh;
      const v8h l0 = *(const v8h*)(lp);        // features 32kt + 8dh + (0..7)
      const v8h l1 = *(const v8h*)(lp + 16);   // features 32kt + 16 + 8dh + (0..7)
      const v16h lv = __builtin_shufflevector(l0, l1, 0, 1, 2, 3, 4, 5, 6, 7,
                                              8, 9, 10, 11, 12, 13, 14, 15);
      const v16h bf = ps[kt] + lv;             // v_pk_add_f16 x8

      // software-rotated A fragment (double buffer)
      v16h a = *(const v16h*)(Wl + ((kt * kFT + 0) * 32 + lane) * 16);
#pragma unroll
      for (int ft = 0; ft < kFT; ++ft) {
        v16h an = a;
        if (ft + 1 < kFT)
          an = *(const v16h*)(Wl + ((kt * kFT + ft + 1) * 32 + lane) * 16);
        acc[ft] = __builtin_amdgcn_wmma_f32_16x16x32_f16(
            false, a, false, bf, (short)0, acc[ft], false, false);
        a = an;
      }
      sched_ds_wmma_interleave();   // keep ds_load pairs flowing between WMMAs
    }

    // e = tanh(acc); LayerNorm over the 256 features of column m
    float s = 0.0f, q = 0.0f;
#pragma unroll
    for (int ft = 0; ft < kFT; ++ft)
#pragma unroll
      for (int v = 0; v < 8; ++v) {
        float e = fast_tanh(acc[ft][v]);
        acc[ft][v] = e;
        s += e;
        q += e * e;
      }
    s += __shfl_xor(s, 16, 32);   // combine the two feature halves of column m
    q += __shfl_xor(q, 16, 32);
    const float mu   = s * (1.0f / kH);
    const float var  = q * (1.0f / kH) - mu * mu;
    const float rstd = rsqrtf(var + kEps);

    // normalize, apply f16 gamma/beta, pack straight into the f16 state (B order)
#pragma unroll
    for (int kt = 0; kt < kKT; ++kt) {
#pragma unroll
      for (int h2 = 0; h2 < 2; ++h2) {
        const int ft = 2 * kt + h2;
        const v8h gv = *(const v8h*)(glh + ft * 16 + 8 * dh);
        const v8h bv = *(const v8h*)(blh + ft * 16 + 8 * dh);
#pragma unroll
        for (int v = 0; v < 8; ++v) {
          const h16 n = (h16)((acc[ft][v] - mu) * rstd);
          ps[kt][8 * h2 + v] = n * gv[v] + bv[v];
        }
      }
    }

    xv = xv_next;
  }

  // readout from registers: out[rowBase+m][o] = sum_f h[f]*out_w[o][f] + out_b[o]
#pragma unroll
  for (int o = 0; o < kOut; ++o) {
    float p = 0.0f;
#pragma unroll
    for (int kt = 0; kt < kKT; ++kt) {
      const v8f ov0 = *(const v8f*)(owl + o * kH + kt * 32 + 8 * dh);
      const v8f ov1 = *(const v8f*)(owl + o * kH + kt * 32 + 16 + 8 * dh);
#pragma unroll
      for (int v = 0; v < 8; ++v) p += (float)ps[kt][v] * ov0[v];
#pragma unroll
      for (int v = 0; v < 8; ++v) p += (float)ps[kt][8 + v] * ov1[v];
    }
    p += __shfl_xor(p, 16, 32);
    if (dh == 0)
      out[(size_t)(rowBase + m) * kOut + o] = p + out_b[o];
  }
}

extern "C" void kernel_launch(void* const* d_in, const int* in_sizes, int n_in,
                              void* d_out, int out_size, void* d_ws, size_t ws_size,
                              hipStream_t stream) {
  (void)in_sizes; (void)n_in; (void)d_ws; (void)ws_size; (void)out_size;
  const float* x   = (const float*)d_in[0];
  const float* ew  = (const float*)d_in[1];
  const float* eb  = (const float*)d_in[2];
  const float* uw  = (const float*)d_in[3];
  const float* ub  = (const float*)d_in[4];
  const float* gm  = (const float*)d_in[5];
  const float* bt  = (const float*)d_in[6];
  const float* ow  = (const float*)d_in[7];
  const float* ob  = (const float*)d_in[8];
  float* out = (float*)d_out;

  (void)hipFuncSetAttribute(reinterpret_cast<const void*>(rnn_scan_wmma),
                            hipFuncAttributeMaxDynamicSharedMemorySize, kSmemBytes);

  dim3 grid(kB / kBRows);   // 256 blocks; 148KB LDS -> up to 2 blocks per WGP
  dim3 block(kBlock);       // 4 waves of 32
  rnn_scan_wmma<<<grid, block, kSmemBytes, stream>>>(x, ew, eb, uw, ub, gm, bt, ow, ob, out);
}
